// Mamba_35966056136769
// MI455X (gfx1250) — matrix operand order repaired and checked by reading, not compiled, not run on hardware
//
#include <hip/hip_runtime.h>
#include <math.h>

// ---------------------------------------------------------------------------
// Mamba time-series forecaster for MI455X (gfx1250, wave32, WMMA).
// Big GEMMs: bf16 v_wmma_f32_16x16x32_bf16, f32 accumulate,
// 2x2 register tiling (32x32 C per wave) + double-buffered K pipeline.
// ---------------------------------------------------------------------------

typedef unsigned short u16;
typedef __attribute__((ext_vector_type(16))) __bf16 v16bf;
typedef __attribute__((ext_vector_type(8)))  float  v8f;

#define BB       32
#define LL       512
#define ENC_IN   8
#define MARKD    4
#define DMODEL   512
#define DSTATE   16
#define DCONV    4
#define DINNER   1024
#define DTRANK   32
#define PREDLEN  96

__device__ __forceinline__ u16 f2bf(float f) {
    unsigned int u = __float_as_uint(f);
    unsigned int r = u + 0x7FFFu + ((u >> 16) & 1u);   // round-to-nearest-even
    return (u16)(r >> 16);
}

// ---------------------------------------------------------------------------
// 1. Per-(b,channel) mean / std over L
// ---------------------------------------------------------------------------
__global__ void k_stats(const float* __restrict__ xe,
                        float* __restrict__ mean, float* __restrict__ stdv) {
    int bc = blockIdx.x;              // b*8 + c
    int b = bc >> 3, c = bc & 7;
    int tid = threadIdx.x;
    float s = 0.f, s2 = 0.f;
    for (int l = tid; l < LL; l += 256) {
        float v = xe[((size_t)b * LL + l) * ENC_IN + c];
        s += v; s2 += v * v;
    }
    __shared__ float rs[256], rs2[256];
    rs[tid] = s; rs2[tid] = s2;
    __syncthreads();
    for (int w = 128; w > 0; w >>= 1) {
        if (tid < w) { rs[tid] += rs[tid + w]; rs2[tid] += rs2[tid + w]; }
        __syncthreads();
    }
    if (tid == 0) {
        float m = rs[0] * (1.f / LL);
        float var = rs2[0] * (1.f / LL) - m * m;
        mean[bc] = m;
        stdv[bc] = sqrtf(var + 1e-5f);
    }
}

// ---------------------------------------------------------------------------
// 2. Embedding: circular token conv + temporal embed + positional encoding,
//    output directly as bf16 (GEMM A operand).  One block per (b,l).
// ---------------------------------------------------------------------------
__global__ void k_embed(const float* __restrict__ xe, const float* __restrict__ xm,
                        const float* __restrict__ cw, const float* __restrict__ tw,
                        const float* __restrict__ mean, const float* __restrict__ stdv,
                        u16* __restrict__ xbf) {
    int bl = blockIdx.x;
    int b = bl >> 9, l = bl & 511;
    __shared__ float sh[28];          // 24 normalized inputs + 4 marks
    int tid = threadIdx.x;
    if (tid < 24) {
        int rr = tid >> 3, c = tid & 7;
        int ls = l + rr - 1;
        ls = (ls < 0) ? (LL - 1) : ((ls > LL - 1) ? 0 : ls);   // circular pad
        sh[tid] = (xe[((size_t)b * LL + ls) * ENC_IN + c] - mean[b * 8 + c])
                  / stdv[b * 8 + c];
    } else if (tid < 28) {
        sh[tid] = xm[((size_t)b * LL + l) * MARKD + (tid - 24)];
    }
    __syncthreads();
    for (int d = tid; d < DMODEL; d += 256) {
        float acc = 0.f;
        const float* w = cw + (size_t)d * 24;       // conv_w[d, c, t]
        #pragma unroll
        for (int c = 0; c < 8; c++)
            #pragma unroll
            for (int t = 0; t < 3; t++)
                acc += sh[t * 8 + c] * w[c * 3 + t];
        #pragma unroll
        for (int c = 0; c < 4; c++) acc += sh[24 + c] * tw[d * 4 + c];
        int j = d >> 1;                              // positional encoding
        float ang = (float)l * __expf(-0.0179889453f * (float)(2 * j)); // ln(1e4)/512
        acc += (d & 1) ? __cosf(ang) : __sinf(ang);
        xbf[(size_t)bl * DMODEL + d] = f2bf(acc);
    }
}

// ---------------------------------------------------------------------------
// 3. bf16 WMMA GEMM:  C[M,N] (f32) = A[M,K] (bf16) * W[N,K]^T (bf16)
//    One wave per 32x32 C block (2x2 WMMA tiles), K stepped by 32,
//    double-buffered so next-step loads fly under current-step WMMAs.
// ---------------------------------------------------------------------------
struct Frag { uint4 q0, q1; };

__device__ __forceinline__ void ldA(Frag& f, const u16* p, int kk, int g) {
    // A: lane holds row M=r; halves 0..7 -> K=kk+g*8+e, 8..15 -> K=kk+16+g*8+e
    f.q0 = *(const uint4*)(p + kk + g * 8);
    f.q1 = *(const uint4*)(p + kk + 16 + g * 8);
}
__device__ __forceinline__ void ldB(Frag& f, const u16* p, int kk, int g) {
    // B (=W^T): lane holds col N=r; halves 0..7 -> K=kk+g*8+e, 8..15 -> K=kk+16+g*8+e
    f.q0 = *(const uint4*)(p + kk + g * 8);
    f.q1 = *(const uint4*)(p + kk + 16 + g * 8);
}
__device__ __forceinline__ v8f wmma_bf16(const Frag& a, const Frag& b, v8f c) {
    union U { Frag f; v16bf v; } ua, ub;
    ua.f = a; ub.f = b;
    return __builtin_amdgcn_wmma_f32_16x16x32_bf16(
               false, ua.v, false, ub.v, (short)0, c, false, false);
}

__global__ void k_gemm_bf16(const u16* __restrict__ A, const u16* __restrict__ W,
                            float* __restrict__ C, int M, int N, int K) {
    int wid = blockIdx.x * 8 + (threadIdx.x >> 5);
    int tiles_n = N >> 5;                     // 32-wide N blocks
    int total = (M >> 5) * tiles_n;
    if (wid >= total) return;                 // wave-uniform: EXEC stays all-1
    int tm = (wid / tiles_n) << 5;
    int tn = (wid % tiles_n) << 5;
    int lane = threadIdx.x & 31;
    int g = lane >> 4, r = lane & 15;

    const u16* pa0 = A + (size_t)(tm + r) * K;
    const u16* pa1 = A + (size_t)(tm + 16 + r) * K;
    const u16* pb0 = W + (size_t)(tn + r) * K;
    const u16* pb1 = W + (size_t)(tn + 16 + r) * K;

    v8f c00 = {}, c01 = {}, c10 = {}, c11 = {};
    Frag a0c, a1c, b0c, b1c, a0n, a1n, b0n, b1n;
    ldA(a0c, pa0, 0, g); ldA(a1c, pa1, 0, g);
    ldB(b0c, pb0, 0, g); ldB(b1c, pb1, 0, g);

    for (int kk = 0; kk < K; kk += 32) {
        int kn = kk + 32;
        if (kn + 32 <= K) {                   // prefetch K step after next
            __builtin_prefetch(pa0 + kn + 32, 0, 0);
            __builtin_prefetch(pb0 + kn + 32, 0, 0);
        }
        int kl = (kn < K) ? kn : 0;           // branch-free tail (redundant ld)
        ldA(a0n, pa0, kl, g); ldA(a1n, pa1, kl, g);
        ldB(b0n, pb0, kl, g); ldB(b1n, pb1, kl, g);
        c00 = wmma_bf16(a0c, b0c, c00);
        c01 = wmma_bf16(a0c, b1c, c01);
        c10 = wmma_bf16(a1c, b0c, c10);
        c11 = wmma_bf16(a1c, b1c, c11);
        a0c = a0n; a1c = a1n; b0c = b0n; b1c = b1n;
    }

    // C/D layout: lane group g holds rows (tile_m + g*8 + v), column tile_n + r
    float* p00 = C + (size_t)(tm + g * 8) * N + tn + r;
    float* p01 = p00 + 16;
    float* p10 = C + (size_t)(tm + 16 + g * 8) * N + tn + r;
    float* p11 = p10 + 16;
    #pragma unroll
    for (int v = 0; v < 8; v++) {
        p00[(size_t)v * N] = c00[v];
        p01[(size_t)v * N] = c01[v];
        p10[(size_t)v * N] = c10[v];
        p11[(size_t)v * N] = c11[v];
    }
}

// ---------------------------------------------------------------------------
// 4. Depthwise causal conv1d + bias + SiLU  ->  u (f32) and u (bf16)
// ---------------------------------------------------------------------------
__global__ void k_dwconv(const float* __restrict__ xz, const float* __restrict__ cw,
                         const float* __restrict__ cb,
                         float* __restrict__ u, u16* __restrict__ ubf) {
    size_t idx = (size_t)blockIdx.x * 256 + threadIdx.x;   // (b<<19)|(l<<10)|d
    int d = (int)(idx & 1023);
    int l = (int)((idx >> 10) & 511);
    int b = (int)(idx >> 19);
    float acc = cb[d];
    #pragma unroll
    for (int t = 0; t < DCONV; t++) {
        int ls = l - (DCONV - 1) + t;
        if (ls >= 0)
            acc += cw[d * DCONV + t] * xz[((size_t)b * LL + ls) * (2 * DINNER) + d];
    }
    float s = acc / (1.f + __expf(-acc));                  // SiLU
    u[idx] = s;
    ubf[idx] = f2bf(s);
}

// ---------------------------------------------------------------------------
// 5. Generic f32 -> bf16 convert, and dt_in slice of x_dbl
// ---------------------------------------------------------------------------
__global__ void k_cvt(const float* __restrict__ in, u16* __restrict__ out, int n) {
    int i = blockIdx.x * 256 + threadIdx.x;
    if (i < n) out[i] = f2bf(in[i]);
}

__global__ void k_slice32(const float* __restrict__ xdbl, u16* __restrict__ dtin) {
    int i = blockIdx.x * 256 + threadIdx.x;   // < 16384*32
    int r = i >> 5, c = i & 31;
    dtin[i] = f2bf(xdbl[(size_t)r * 64 + c]);
}

// ---------------------------------------------------------------------------
// 6. Selective scan. One thread per (b,d); h[16] in registers; B/C in LDS.
//    Emits y (bf16) only for the final 96 timesteps (all out_proj needs).
// ---------------------------------------------------------------------------
__global__ void k_scan(const float* __restrict__ dtr, const float* __restrict__ dtb,
                       const float* __restrict__ u, const float* __restrict__ xdbl,
                       const float* __restrict__ xz, const float* __restrict__ A_log,
                       const float* __restrict__ Dp, u16* __restrict__ ylast) {
    int b = blockIdx.x >> 2;
    int d = ((blockIdx.x & 3) << 8) + threadIdx.x;
    float A[DSTATE], h[DSTATE];
    #pragma unroll
    for (int s = 0; s < DSTATE; s++) {
        A[s] = -__expf(A_log[(size_t)d * DSTATE + s]);
        h[s] = 0.f;
    }
    float bias = dtb[d], Dd = Dp[d];
    __shared__ float sBC[32];                  // [0:16)=B_t, [16:32)=C_t
    for (int l = 0; l < LL; l++) {
        size_t row = (size_t)b * LL + l;
        __syncthreads();
        if (threadIdx.x < 32)
            sBC[threadIdx.x] = xdbl[row * 64 + DTRANK + threadIdx.x];
        __syncthreads();
        float dtv = dtr[row * DINNER + d] + bias;
        dtv = (dtv > 20.f) ? dtv : log1pf(__expf(dtv));    // softplus
        float uv = u[row * DINNER + d];
        float du = dtv * uv;
        float y = 0.f;
        #pragma unroll
        for (int s = 0; s < DSTATE; s++) {
            h[s] = __expf(dtv * A[s]) * h[s] + du * sBC[s];
            y += h[s] * sBC[16 + s];
        }
        y += uv * Dd;
        float z = xz[row * (2 * DINNER) + DINNER + d];
        y *= z / (1.f + __expf(-z));                       // * silu(z)
        if (l >= LL - PREDLEN)
            ylast[((size_t)b * PREDLEN + (l - (LL - PREDLEN))) * DINNER + d] = f2bf(y);
    }
}

// ---------------------------------------------------------------------------
// 7. Final head: 512-dot with out_w + de-normalization
// ---------------------------------------------------------------------------
__global__ void k_out(const float* __restrict__ mout, const float* __restrict__ ow,
                      const float* __restrict__ mean, const float* __restrict__ stdv,
                      float* __restrict__ out) {
    int row = blockIdx.x;                      // b*96 + t
    int b = row / PREDLEN;
    int tid = threadIdx.x;
    float s = mout[(size_t)row * DMODEL + tid] * ow[tid]
            + mout[(size_t)row * DMODEL + 256 + tid] * ow[256 + tid];
    __shared__ float rs[256];
    rs[tid] = s; __syncthreads();
    for (int w = 128; w > 0; w >>= 1) {
        if (tid < w) rs[tid] += rs[tid + w];
        __syncthreads();
    }
    if (tid == 0) out[row] = rs[0] * stdv[b * 8] + mean[b * 8];
}

// ---------------------------------------------------------------------------
// Host: workspace carve-up + launch schedule (all on `stream`, graph-safe)
// ---------------------------------------------------------------------------
extern "C" void kernel_launch(void* const* d_in, const int* in_sizes, int n_in,
                              void* d_out, int out_size, void* d_ws, size_t ws_size,
                              hipStream_t stream) {
    const float* x_enc    = (const float*)d_in[0];
    const float* x_mark   = (const float*)d_in[1];
    const float* conv_w   = (const float*)d_in[2];
    const float* temp_w   = (const float*)d_in[3];
    const float* in_proj  = (const float*)d_in[4];
    const float* c1d_w    = (const float*)d_in[5];
    const float* c1d_b    = (const float*)d_in[6];
    const float* xproj_w  = (const float*)d_in[7];
    const float* dtproj_w = (const float*)d_in[8];
    const float* dtproj_b = (const float*)d_in[9];
    const float* A_log    = (const float*)d_in[10];
    const float* Dp       = (const float*)d_in[11];
    const float* outproj_w= (const float*)d_in[12];
    const float* out_w    = (const float*)d_in[13];
    float* out = (float*)d_out;

    char* w = (char*)d_ws;
    size_t off = 0;
    auto alloc = [&](size_t bytes) -> void* {
        void* p = w + off;
        off += (bytes + 255) & ~(size_t)255;
        return p;
    };
    const size_t ML  = (size_t)BB * LL;             // 16384 rows
    float* mean  = (float*)alloc(256 * 4);
    float* stdv  = (float*)alloc(256 * 4);
    u16*   xbf   = (u16*)  alloc(ML * DMODEL * 2);          // embed, bf16
    u16*   w_in  = (u16*)  alloc((size_t)2 * DINNER * DMODEL * 2);
    u16*   w_xp  = (u16*)  alloc((size_t)64 * DINNER * 2);
    u16*   w_dt  = (u16*)  alloc((size_t)DINNER * DTRANK * 2);
    u16*   w_op  = (u16*)  alloc((size_t)DMODEL * DINNER * 2);
    float* xz    = (float*)alloc(ML * 2 * DINNER * 4);      // 134 MB
    float* uf    = (float*)alloc(ML * DINNER * 4);
    u16*   ubf   = (u16*)  alloc(ML * DINNER * 2);
    float* xdbl  = (float*)alloc(ML * 64 * 4);
    u16*   dtin  = (u16*)  alloc(ML * DTRANK * 2);
    float* dtraw = (float*)alloc(ML * DINNER * 4);
    u16*   ylast = (u16*)  alloc((size_t)BB * PREDLEN * DINNER * 2);
    float* mout  = (float*)alloc((size_t)BB * PREDLEN * DMODEL * 4);
    (void)ws_size; (void)in_sizes; (void)n_in; (void)out_size;

    // 1. stats
    k_stats<<<BB * ENC_IN, 256, 0, stream>>>(x_enc, mean, stdv);

    // 2. weight conversions to bf16
    int n1 = 2 * DINNER * DMODEL;   // 1,048,576
    int n2 = 64 * DINNER;           // 65,536
    int n3 = DINNER * DTRANK;       // 32,768
    int n4 = DMODEL * DINNER;       // 524,288
    k_cvt<<<(n1 + 255) / 256, 256, 0, stream>>>(in_proj,   w_in, n1);
    k_cvt<<<(n2 + 255) / 256, 256, 0, stream>>>(xproj_w,   w_xp, n2);
    k_cvt<<<(n3 + 255) / 256, 256, 0, stream>>>(dtproj_w,  w_dt, n3);
    k_cvt<<<(n4 + 255) / 256, 256, 0, stream>>>(outproj_w, w_op, n4);

    // 3. embedding (bf16 A operand)
    k_embed<<<BB * LL, 256, 0, stream>>>(x_enc, x_mark, conv_w, temp_w,
                                         mean, stdv, xbf);

    // 4. in_proj: [16384,512] x [2048,512]^T -> xz
    //    32x32 blocks: (16384/32)*(2048/32) = 32768 waves -> 4096 blocks
    k_gemm_bf16<<<4096, 256, 0, stream>>>(xbf, w_in, xz,
                                          (int)ML, 2 * DINNER, DMODEL);

    // 5. depthwise conv + SiLU -> u
    k_dwconv<<<(int)(ML * DINNER / 256), 256, 0, stream>>>(xz, c1d_w, c1d_b,
                                                           uf, ubf);

    // 6. x_proj: [16384,1024] x [64,1024]^T -> x_dbl  (1024 waves -> 128 blocks)
    k_gemm_bf16<<<128, 256, 0, stream>>>(ubf, w_xp, xdbl, (int)ML, 64, DINNER);

    // 7. dt_in slice -> bf16
    k_slice32<<<(int)(ML * DTRANK / 256), 256, 0, stream>>>(xdbl, dtin);

    // 8. dt_proj: [16384,32] x [1024,32]^T -> dt_raw  (16384 waves -> 2048 blocks)
    k_gemm_bf16<<<2048, 256, 0, stream>>>(dtin, w_dt, dtraw,
                                          (int)ML, DINNER, DTRANK);

    // 9. selective scan (+ softplus, gate, D skip); emits last-96 y as bf16
    k_scan<<<BB * 4, 256, 0, stream>>>(dtraw, dtproj_b, uf, xdbl, xz,
                                       A_log, Dp, ylast);

    // 10. out_proj on last 96 steps only: [3072,1024] x [512,1024]^T
    //     1536 waves -> 192 blocks
    k_gemm_bf16<<<192, 256, 0, stream>>>(ylast, w_op, mout,
                                         BB * PREDLEN, DMODEL, DINNER);

    // 11. head + de-normalization
    k_out<<<BB * PREDLEN, 256, 0, stream>>>(mout, out_w, mean, stdv, out);
}
